// RNN_37271726195326
// MI455X (gfx1250) — matrix operand order repaired and checked by reading, not compiled, not run on hardware
//
#include <hip/hip_runtime.h>

typedef __attribute__((ext_vector_type(16))) _Float16 v16h;
typedef __attribute__((ext_vector_type(8)))  _Float16 v8h;
typedef __attribute__((ext_vector_type(8)))  float    v8f;

#define B_SZ 2048
#define T_SZ 128
#define F_SZ 128
#define U2N  400      // 4 * 100 gate width
#define KP   136      // padded K stride in f16 elems (100 h + 4 h1-in + pad), 272B row

__device__ __forceinline__ float sigf(float x) { return 1.0f / (1.0f + __expf(-x)); }

// ---------------- K0: zx1[r][16] = x[r][:] @ W1 + b1, r = b*T + t -----------------
__global__ void __launch_bounds__(256) k0_inproj(const float* __restrict__ x,
                                                 const float* __restrict__ W1,
                                                 const float* __restrict__ b1,
                                                 float* __restrict__ zx1) {
    __shared__ float sW[F_SZ * 16];
    __shared__ float sb[16];
    int tid = threadIdx.x;
    for (int i = tid; i < F_SZ * 16; i += 256) sW[i] = W1[i];
    if (tid < 16) sb[tid] = b1[tid];
    __syncthreads();

    long r = (long)blockIdx.x * 256 + tid;
    const float4* x4 = (const float4*)(x + r * F_SZ);
    float acc[16];
#pragma unroll
    for (int g = 0; g < 16; ++g) acc[g] = sb[g];
    for (int f4 = 0; f4 < F_SZ / 4; ++f4) {
        float4 xv = x4[f4];
        const float* w = sW + f4 * 4 * 16;
#pragma unroll
        for (int g = 0; g < 16; ++g) acc[g] += xv.x * w[g];
#pragma unroll
        for (int g = 0; g < 16; ++g) acc[g] += xv.y * w[16 + g];
#pragma unroll
        for (int g = 0; g < 16; ++g) acc[g] += xv.z * w[32 + g];
#pragma unroll
        for (int g = 0; g < 16; ++g) acc[g] += xv.w * w[48 + g];
    }
    float4* o4 = (float4*)(zx1 + r * 16);
#pragma unroll
    for (int q = 0; q < 4; ++q)
        o4[q] = make_float4(acc[q*4], acc[q*4+1], acc[q*4+2], acc[q*4+3]);
}

// ---------------- K1: LSTM1 recurrence (units=4), one thread per batch row --------
__global__ void __launch_bounds__(256) k1_lstm1(const float* __restrict__ zx1,
                                                const float* __restrict__ U1,
                                                float* __restrict__ h1out) {
    __shared__ float sU1[64];
    int tid = threadIdx.x;
    if (tid < 64) sU1[tid] = U1[tid];
    __syncthreads();

    int b = blockIdx.x * 256 + tid;
    float h[4] = {0.f,0.f,0.f,0.f}, c[4] = {0.f,0.f,0.f,0.f};
    const float4* z4 = (const float4*)(zx1 + (long)b * T_SZ * 16);
    for (int t = 0; t < T_SZ; ++t) {
        float p[16];
#pragma unroll
        for (int q = 0; q < 4; ++q) {
            float4 v = z4[t * 4 + q];
            p[q*4] = v.x; p[q*4+1] = v.y; p[q*4+2] = v.z; p[q*4+3] = v.w;
        }
#pragma unroll
        for (int g = 0; g < 16; ++g)
            p[g] += h[0]*sU1[g] + h[1]*sU1[16+g] + h[2]*sU1[32+g] + h[3]*sU1[48+g];
#pragma unroll
        for (int j = 0; j < 4; ++j) {
            float ig = sigf(p[j]);
            float fg = sigf(p[4 + j]);
            float gg = fmaxf(p[8 + j], 0.f);
            float og = sigf(p[12 + j]);
            c[j] = fg * c[j] + ig * gg;
            h[j] = og * fmaxf(c[j], 0.f);
        }
        *(float4*)(h1out + ((long)t * B_SZ + b) * 4) = make_float4(h[0], h[1], h[2], h[3]);
    }
}

// ---------------- K2: LSTM2 recurrence with WMMA, 16 batch rows per block ---------
// A (LDS sA): [m=16][k=KP] f16; k<100 = h2 state, k=100..103 = h1 inputs, rest 0.
// B: U2ext[n][k] f16 (k<100 = U2[k][n], 100..103 = W2[k-100][n]); staged once in LDS,
//    then held in per-wave REGISTERS (v16h bfr[4][4]) for all 128 timesteps.
// h1 inputs are software-pipelined: h1(t+1) prefetched during step t.
__global__ void __launch_bounds__(256) k2_lstm2(const float* __restrict__ h1,
                                                const float* __restrict__ W2,
                                                const float* __restrict__ U2,
                                                const float* __restrict__ b2,
                                                float* __restrict__ h2out) {
    __shared__ _Float16 sU[U2N * KP];   // 108800 B (staging; dead after setup)
    __shared__ _Float16 sA[16 * KP];    //   4352 B
    __shared__ float    sZ[16 * U2N];   //  25600 B
    __shared__ float    sB2[U2N];       //   1600 B

    int tid  = threadIdx.x;
    int lane = tid & 31;
    // wave id as a true scalar so tile-selection branches are SCC-based (EXEC stays all-1)
    int wv   = __builtin_amdgcn_readfirstlane(tid >> 5);
    int brow0 = blockIdx.x * 16;

    // Build extended B image (coalesced global reads, scattered LDS writes).
    for (int idx = tid; idx < 104 * U2N; idx += 256) {
        int k = idx / U2N, n = idx - k * U2N;
        float v = (k < 100) ? U2[idx] : W2[idx - 100 * U2N];
        sU[n * KP + k] = (_Float16)v;
    }
    for (int idx = tid; idx < U2N * (KP - 104); idx += 256) {
        int n = idx / (KP - 104), k = 104 + (idx - n * (KP - 104));
        sU[n * KP + k] = (_Float16)0.f;
    }
    for (int n = tid; n < U2N; n += 256) sB2[n] = b2[n];
    for (int i = tid; i < 16 * KP; i += 256) sA[i] = (_Float16)0.f;

    // Pointwise ownership: 1600 (m,u) elements round-robined over 256 threads.
    float cst[7];
    int epk[7];                     // (m<<8)|u
    int cnt = 0;
    for (int e = tid; e < 1600; e += 256) {
        int m = e / 100;
        epk[cnt] = (m << 8) | (e - m * 100);
        cst[cnt] = 0.f;
        ++cnt;
    }

    // h1 input staging: thread tid<64 owns element (m = tid>>2, j = tid&3).
    int h1_m = tid >> 2, h1_j = tid & 3;
    const float* h1p = h1 + (long)(brow0 + h1_m) * 4 + h1_j;  // + t*B_SZ*4 per step
    int   h1_lds = h1_m * KP + 100 + h1_j;
    float h1reg = 0.f;
    if (tid < 64) h1reg = h1p[0];                              // prefetch t = 0

    // Fragment addressing (ISA 16-bit A / B layouts, wave32).
    const _Float16* aBase = sA + (lane & 15) * KP + (lane >> 4) * 8;
    const int nB_m    = lane & 15;
    const int nB_koff = (lane >> 4) * 8;
    const int c_mbase = (lane >> 4) * 8;
    const int ntc     = (wv == 0) ? 4 : 3;   // tiles per wave: wv, wv+8, wv+16[, wv+24]

    __syncthreads();

    // Hoist this wave's B fragments into registers for the whole time loop.
    v16h bfr[4][4];
    float bvv[4];
    int   zbase[4];
#pragma unroll
    for (int i = 0; i < 4; ++i) {
        int nt = wv + 8 * i;
        int n  = nt * 16 + nB_m;
        if (i < ntc) {
            bvv[i]   = sB2[n];
            zbase[i] = c_mbase * U2N + n;
#pragma unroll
            for (int kt = 0; kt < 4; ++kt) {
                const _Float16* bp = sU + n * KP + kt * 32 + nB_koff;
                v8h lo = *(const v8h*)bp;
                v8h hi = *(const v8h*)(bp + 16);
                v16h bb;
#pragma unroll
                for (int q = 0; q < 8; ++q) { bb[q] = lo[q]; bb[q + 8] = hi[q]; }
                bfr[i][kt] = bb;
            }
        } else {
            bvv[i] = 0.f; zbase[i] = 0;
#pragma unroll
            for (int kt = 0; kt < 4; ++kt) bfr[i][kt] = v16h{};
        }
    }

    for (int t = 0; t < T_SZ; ++t) {
        if (tid < 64) {
            sA[h1_lds] = (_Float16)h1reg;            // publish h1(t)
            if (t + 1 < T_SZ)                         // prefetch h1(t+1): latency
                h1reg = h1p[(long)(t + 1) * B_SZ * 4];//   hidden behind WMMA phase
        }
        __syncthreads();   // sA (state + inputs) ready; sZ free to overwrite

        v16h afr[4];
#pragma unroll
        for (int kt = 0; kt < 4; ++kt) {
            const _Float16* ap = aBase + kt * 32;
            v8h lo = *(const v8h*)ap;          // K base..base+7
            v8h hi = *(const v8h*)(ap + 16);   // K base+16..base+23
            v16h a;
#pragma unroll
            for (int q = 0; q < 8; ++q) { a[q] = lo[q]; a[q + 8] = hi[q]; }
            afr[kt] = a;
        }

#pragma unroll
        for (int i = 0; i < 4; ++i) {
            if (i < ntc) {                     // scalar (SCC) branch, EXEC all-1
                v8f acc = {};
#pragma unroll
                for (int kt = 0; kt < 4; ++kt)
                    acc = __builtin_amdgcn_wmma_f32_16x16x32_f16(
                              false, afr[kt], false, bfr[i][kt], (short)0, acc,
                              false, false);
                float bv = bvv[i];
#pragma unroll
                for (int j = 0; j < 8; ++j)
                    sZ[zbase[i] + j * U2N] = acc[j] + bv;
            }
        }
        __syncthreads();   // sZ complete

        // Gates + state update (Keras order i,f,g,o; act=relu, rec_act=sigmoid)
#pragma unroll
        for (int q = 0; q < 7; ++q) {
            if (q < cnt) {
                int m = epk[q] >> 8, u = epk[q] & 255;
                const float* zr = sZ + m * U2N + u;
                float ig = sigf(zr[0]);
                float fg = sigf(zr[100]);
                float gg = fmaxf(zr[200], 0.f);
                float og = sigf(zr[300]);
                float cc = fg * cst[q] + ig * gg;
                cst[q] = cc;
                float hh = og * fmaxf(cc, 0.f);
                sA[m * KP + u] = (_Float16)hh;   // next-step A operand
                if (t == T_SZ - 1)
                    h2out[(long)(brow0 + m) * 100 + u] = hh;
            }
        }
        // no barrier here: next-iter barrier orders sA writes vs WMMA reads
    }
}

// ---------------- K3: dense(36, relu) -> dense(1, sigmoid) ------------------------
__global__ void __launch_bounds__(256) k3_dense(const float* __restrict__ h2,
                                                const float* __restrict__ Wd,
                                                const float* __restrict__ bd,
                                                const float* __restrict__ Wo,
                                                const float* __restrict__ bo,
                                                float* __restrict__ out) {
    __shared__ float sWd[100 * 36];
    __shared__ float sbd[36], sWo[36];
    __shared__ float sbo;
    int tid = threadIdx.x;
    for (int i = tid; i < 3600; i += 256) sWd[i] = Wd[i];
    if (tid < 36) { sbd[tid] = bd[tid]; sWo[tid] = Wo[tid]; }
    if (tid == 0) sbo = bo[0];
    __syncthreads();

    int b = blockIdx.x * 256 + tid;
    const float* h = h2 + (long)b * 100;
    float d[36];
#pragma unroll
    for (int j = 0; j < 36; ++j) d[j] = sbd[j];
    for (int u = 0; u < 100; ++u) {
        float hu = h[u];
#pragma unroll
        for (int j = 0; j < 36; ++j) d[j] += hu * sWd[u * 36 + j];
    }
    float acc = sbo;
#pragma unroll
    for (int j = 0; j < 36; ++j) acc += fmaxf(d[j], 0.f) * sWo[j];
    out[b] = sigf(acc);
}

extern "C" void kernel_launch(void* const* d_in, const int* in_sizes, int n_in,
                              void* d_out, int out_size, void* d_ws, size_t ws_size,
                              hipStream_t stream) {
    const float* x  = (const float*)d_in[0];
    const float* W1 = (const float*)d_in[1];
    const float* U1 = (const float*)d_in[2];
    const float* b1 = (const float*)d_in[3];
    const float* W2 = (const float*)d_in[4];
    const float* U2 = (const float*)d_in[5];
    const float* b2 = (const float*)d_in[6];
    const float* Wd = (const float*)d_in[7];
    const float* bd = (const float*)d_in[8];
    const float* Wo = (const float*)d_in[9];
    const float* bo = (const float*)d_in[10];
    float* out = (float*)d_out;

    float* zx1 = (float*)d_ws;                           // B*T*16 f32 = 16 MB
    float* h1  = zx1 + (size_t)B_SZ * T_SZ * 16;         // T*B*4  f32 =  4 MB
    float* h2  = h1  + (size_t)T_SZ * B_SZ * 4;          // B*100  f32 = 0.8 MB

    k0_inproj<<<(B_SZ * T_SZ) / 256, 256, 0, stream>>>(x, W1, b1, zx1);
    k1_lstm1 <<<B_SZ / 256,        256, 0, stream>>>(zx1, U1, h1);
    k2_lstm2 <<<B_SZ / 16,         256, 0, stream>>>(h1, W2, U2, b2, h2);
    k3_dense <<<B_SZ / 256,        256, 0, stream>>>(h2, Wd, bd, Wo, bo, out);
}
